// InterpretableMultiheadAttention_43671227466294
// MI455X (gfx1250) — matrix operand order repaired and checked by reading, not compiled, not run on hardware
//
#include <hip/hip_runtime.h>
#include <hip/hip_bf16.h>

// ---------------------------------------------------------------------------
// Multihead attention forward (B=4, S=2048, D=1024, H=16, d_k=64) for gfx1250.
// All GEMMs run on v_wmma_f32_16x16x32_f16 (wave32). V is stored transposed so
// every WMMA B-fragment is two 16-byte k-contiguous loads per lane. Flash-style
// online softmax keeps the S x S score matrix out of memory. Projection GEMMs
// use 32x64 tiles per wave (2 A-frags x 4 B-frags -> 8 WMMA per k-step) to
// double weight-fragment reuse.
// ---------------------------------------------------------------------------

typedef __attribute__((ext_vector_type(16))) _Float16 v16h;
typedef __attribute__((ext_vector_type(8)))  _Float16 v8h;
typedef __attribute__((ext_vector_type(8)))  float    v8f;

#define BATCH    4
#define SEQ      2048
#define DMODEL   1024
#define NHEADS   16
#define DK       64
#define NROWS    (BATCH * SEQ)        // 8192
#define QKV_N    (3 * DMODEL)         // 3072

// ---------------------------------------------------------------------------
// WMMA wrapper: D = A(16x32 f16) * B(32x16 f16) + C(16x16 f32)
// ---------------------------------------------------------------------------
__device__ __forceinline__ v8f wmma_f16(v16h a, v16h b, v8f c) {
  return __builtin_amdgcn_wmma_f32_16x16x32_f16(
      /*neg_a=*/false, a, /*neg_b=*/false, b,
      /*c_mod=*/(short)0, c, /*reuse_a=*/false, /*reuse_b=*/false);
}

// A-fragment (16x32, M x K), 16-bit layout (ISA 7.12.2):
//   lanes 0-15: m = lane,   halves 0-7 -> k = kBase+0..7,  halves 8-15 -> k = kBase+16..23
//   lanes 16-31: m = lane-16, same but k offset +8
__device__ __forceinline__ v16h load_a_frag(const _Float16* __restrict__ A,
                                            int lda, int mBase, int kBase) {
  int lane = threadIdx.x & 31;
  int m  = mBase + (lane & 15);
  int kb = kBase + ((lane >> 4) << 3);          // +8 for hi half-wave
  const _Float16* p = A + (size_t)m * lda + kb;
  v16h r;
  *((v8h*)&r)     = *(const v8h*)(p);           // halves 0..7
  *((v8h*)&r + 1) = *(const v8h*)(p + 16);      // halves 8..15
  return r;
}

// B-fragment (32x16, K x N) where memory holds Mat[n][k] row-major (k contig):
//   lane: n = nBase + (lane&15); halves 0-7 -> k = kBase + (lane>=16 ? 8:0) + 0..7,
//   halves 8-15 -> same + 16. Two 16-byte (v8h) loads per lane.
__device__ __forceinline__ v16h load_b_frag(const _Float16* __restrict__ Mat,
                                            int ldk, int nBase, int kBase) {
  int lane = threadIdx.x & 31;
  int n = nBase + (lane & 15);
  int k = kBase + ((lane >> 4) << 3);           // +8 for hi half-wave
  const _Float16* p = Mat + (size_t)n * ldk + k;
  v16h r;
  *((v8h*)&r)     = *(const v8h*)(p);
  *((v8h*)&r + 1) = *(const v8h*)(p + 16);
  return r;
}

// Reductions across a 16-lane group (xor masks 1,2,4,8 never cross bit 4,
// so lo/hi half-wave rows stay independent — matches the C/D VGPR layout).
__device__ __forceinline__ float group_max(float v) {
  #pragma unroll
  for (int off = 1; off < 16; off <<= 1) v = fmaxf(v, __shfl_xor(v, off, 32));
  return v;
}
__device__ __forceinline__ float group_sum(float v) {
  #pragma unroll
  for (int off = 1; off < 16; off <<= 1) v += __shfl_xor(v, off, 32);
  return v;
}

// ---------------------------------------------------------------------------
// Kernel 0: f32 -> f16 conversion
// ---------------------------------------------------------------------------
__global__ void cvt_f32_f16(const float* __restrict__ src,
                            _Float16* __restrict__ dst, int n) {
  int i = blockIdx.x * blockDim.x + threadIdx.x;
  if (i < n) dst[i] = (_Float16)src[i];
}

// ---------------------------------------------------------------------------
// Kernel 1: fused QKV projection.  y = x @ W^T + b, W packed [3072][1024].
// Each wave -> 32 (M) x 64 (N) tile: 2 A-frags reuse 4 B-frags (8 WMMA/k-step).
// Epilogue scatters into Q/K ([bh][s][d]) and V transposed ([bh][d][s]);
// Q is pre-scaled by 1/sqrt(d_k) = 0.125.
// grid = (NROWS/256, QKV_N/64), block = 256 (8 waves)
// ---------------------------------------------------------------------------
__global__ __launch_bounds__(256) void qkv_gemm(
    const _Float16* __restrict__ Xh, const _Float16* __restrict__ Wqkv,
    const float* __restrict__ bq, const float* __restrict__ bk,
    const float* __restrict__ bv,
    _Float16* __restrict__ Qh, _Float16* __restrict__ Kh,
    _Float16* __restrict__ Vt) {
  int wave = threadIdx.x >> 5;
  int lane = threadIdx.x & 31;
  int mBase = (blockIdx.x * 8 + wave) * 32;     // 32 rows per wave
  int nBase = blockIdx.y * 64;

  v8f c[2][4] = {{v8f{}, v8f{}, v8f{}, v8f{}}, {v8f{}, v8f{}, v8f{}, v8f{}}};
  for (int k = 0; k < DMODEL; k += 32) {
    v16h a0 = load_a_frag(Xh, DMODEL, mBase,      k);
    v16h a1 = load_a_frag(Xh, DMODEL, mBase + 16, k);
    #pragma unroll
    for (int t = 0; t < 4; ++t) {
      v16h b = load_b_frag(Wqkv, DMODEL, nBase + t * 16, k);
      c[0][t] = wmma_f16(a0, b, c[0][t]);
      c[1][t] = wmma_f16(a1, b, c[1][t]);
    }
  }

  int which = nBase >> 10;                 // 0=Q 1=K 2=V
  int nn    = nBase & 1023;                // column within the 1024-wide matrix
  int h     = nn >> 6;                     // uniform across the 64-wide N tile
  const float* bias = (which == 0) ? bq : (which == 1) ? bk : bv;
  float qscale = (which == 0) ? 0.125f : 1.0f;   // fold 1/sqrt(64) into Q
  int batch = mBase >> 11;                 // 32-row tile never crosses a batch
  int sBase = mBase & 2047;
  int bh    = batch * NHEADS + h;
  int hiOff = (lane >> 4) << 3;            // +8 rows for hi half-wave

  #pragma unroll
  for (int mi = 0; mi < 2; ++mi) {
    #pragma unroll
    for (int t = 0; t < 4; ++t) {
      int d = t * 16 + (lane & 15);
      float biasv = bias[nn + t * 16 + (lane & 15)];
      #pragma unroll
      for (int r = 0; r < 8; ++r) {
        int s = sBase + mi * 16 + r + hiOff;
        float val = (c[mi][t][r] + biasv) * qscale;
        if (which == 2)
          Vt[((size_t)bh * DK + d) * SEQ + s] = (_Float16)val;
        else if (which == 0)
          Qh[((size_t)bh * SEQ + s) * DK + d] = (_Float16)val;
        else
          Kh[((size_t)bh * SEQ + s) * DK + d] = (_Float16)val;
      }
    }
  }
}

// ---------------------------------------------------------------------------
// Kernel 2: flash attention.  One workgroup = one (b,h) x 128 query rows,
// each wave owns 16 query rows. Keys streamed in blocks of 32 with online
// softmax; P bounced through per-wave LDS to re-layout C-frag -> A-frag.
// grid = (SEQ/128, BATCH*NHEADS), block = 256
// ---------------------------------------------------------------------------
__global__ __launch_bounds__(256) void attn_kernel(
    const _Float16* __restrict__ Qh, const _Float16* __restrict__ Kh,
    const _Float16* __restrict__ Vt, _Float16* __restrict__ Ctx) {
  int wave = threadIdx.x >> 5;
  int lane = threadIdx.x & 31;
  int bh   = blockIdx.y;
  int qBase = blockIdx.x * 128 + wave * 16;

  const _Float16* Q = Qh + (size_t)bh * SEQ * DK;
  const _Float16* K = Kh + (size_t)bh * SEQ * DK;
  const _Float16* V = Vt + (size_t)bh * DK * SEQ;

  __shared__ _Float16 pShared[8][16][32];       // per-wave P tile (8 KB)

  // Q A-fragments: 16 rows x 64 dims = two 16x32 fragments (loaded once).
  v16h qa0 = load_a_frag(Q, DK, qBase, 0);
  v16h qa1 = load_a_frag(Q, DK, qBase, 32);

  v8f O0 = {}, O1 = {}, O2 = {}, O3 = {};
  float mrow[8], lrow[8];
  #pragma unroll
  for (int r = 0; r < 8; ++r) { mrow[r] = -1e30f; lrow[r] = 0.0f; }

  for (int kb = 0; kb < SEQ; kb += 32) {
    // ---- scores S = (Q/sqrt(dk)) . K^T for 16 q-rows x 32 keys -----------
    v8f c0 = {}, c1 = {};
    c0 = wmma_f16(qa0, load_b_frag(K, DK, kb,       0),  c0);
    c0 = wmma_f16(qa1, load_b_frag(K, DK, kb,       32), c0);
    c1 = wmma_f16(qa0, load_b_frag(K, DK, kb + 16,  0),  c1);
    c1 = wmma_f16(qa1, load_b_frag(K, DK, kb + 16,  32), c1);

    // ---- online softmax (row r lives in VGPR r; lanes 0-15 = rows 0-7,
    //      lanes 16-31 = rows 8-15, identical layout to the O accumulator) --
    #pragma unroll
    for (int r = 0; r < 8; ++r) {
      float tmax  = group_max(fmaxf(c0[r], c1[r]));
      float nmax  = fmaxf(mrow[r], tmax);
      float scale = __expf(mrow[r] - nmax);
      float p0 = __expf(c0[r] - nmax);
      float p1 = __expf(c1[r] - nmax);
      float rsum = group_sum(p0 + p1);
      lrow[r] = lrow[r] * scale + rsum;
      mrow[r] = nmax;
      O0[r] *= scale; O1[r] *= scale; O2[r] *= scale; O3[r] *= scale;
      int m = r + ((lane >> 4) << 3);
      int n = lane & 15;
      pShared[wave][m][n]      = (_Float16)p0;
      pShared[wave][m][16 + n] = (_Float16)p1;
    }
    asm volatile("s_wait_dscnt 0" ::: "memory");  // wave-private LDS RAW

    // ---- re-load P as an A-fragment (16 x 32 keys) -----------------------
    v16h pa;
    {
      int m   = lane & 15;
      int kb2 = (lane >> 4) << 3;
      const _Float16* pp = &pShared[wave][m][kb2];
      *((v8h*)&pa)     = *(const v8h*)(pp);
      *((v8h*)&pa + 1) = *(const v8h*)(pp + 16);
    }

    // ---- O += P . V  (V transposed -> contiguous B-fragment loads) -------
    O0 = wmma_f16(pa, load_b_frag(V, SEQ, 0,  kb), O0);
    O1 = wmma_f16(pa, load_b_frag(V, SEQ, 16, kb), O1);
    O2 = wmma_f16(pa, load_b_frag(V, SEQ, 32, kb), O2);
    O3 = wmma_f16(pa, load_b_frag(V, SEQ, 48, kb), O3);
  }

  // ---- finalize: O / l, store ctx as f16 [b*S+s][h*64+d] -----------------
  int batch = bh >> 4;
  int head  = bh & 15;
  int hiOff = (lane >> 4) << 3;
  #pragma unroll
  for (int r = 0; r < 8; ++r) {
    float inv = 1.0f / lrow[r];
    int s = qBase + r + hiOff;
    size_t rowOff = ((size_t)batch * SEQ + s) * DMODEL + head * DK;
    int d = lane & 15;
    Ctx[rowOff + d]      = (_Float16)(O0[r] * inv);
    Ctx[rowOff + 16 + d] = (_Float16)(O1[r] * inv);
    Ctx[rowOff + 32 + d] = (_Float16)(O2[r] * inv);
    Ctx[rowOff + 48 + d] = (_Float16)(O3[r] * inv);
  }
}

// ---------------------------------------------------------------------------
// Kernel 3: output projection.  out = ctx @ Wo^T + bo  (f32 output)
// 32x64 tile per wave, same structure as qkv_gemm.
// grid = (NROWS/256, DMODEL/64), block = 256
// ---------------------------------------------------------------------------
__global__ __launch_bounds__(256) void out_gemm(
    const _Float16* __restrict__ Ctx, const _Float16* __restrict__ Woh,
    const float* __restrict__ bo, float* __restrict__ Out) {
  int wave = threadIdx.x >> 5;
  int lane = threadIdx.x & 31;
  int mBase = (blockIdx.x * 8 + wave) * 32;
  int nBase = blockIdx.y * 64;

  v8f c[2][4] = {{v8f{}, v8f{}, v8f{}, v8f{}}, {v8f{}, v8f{}, v8f{}, v8f{}}};
  for (int k = 0; k < DMODEL; k += 32) {
    v16h a0 = load_a_frag(Ctx, DMODEL, mBase,      k);
    v16h a1 = load_a_frag(Ctx, DMODEL, mBase + 16, k);
    #pragma unroll
    for (int t = 0; t < 4; ++t) {
      v16h b = load_b_frag(Woh, DMODEL, nBase + t * 16, k);
      c[0][t] = wmma_f16(a0, b, c[0][t]);
      c[1][t] = wmma_f16(a1, b, c[1][t]);
    }
  }
  int hiOff = (lane >> 4) << 3;
  #pragma unroll
  for (int mi = 0; mi < 2; ++mi) {
    #pragma unroll
    for (int t = 0; t < 4; ++t) {
      int n = nBase + t * 16 + (lane & 15);
      float biasv = bo[n];
      #pragma unroll
      for (int r = 0; r < 8; ++r) {
        int row = mBase + mi * 16 + r + hiOff;
        Out[(size_t)row * DMODEL + n] = c[mi][t][r] + biasv;
      }
    }
  }
}

// ---------------------------------------------------------------------------
// Host launcher
// ---------------------------------------------------------------------------
extern "C" void kernel_launch(void* const* d_in, const int* in_sizes, int n_in,
                              void* d_out, int out_size, void* d_ws, size_t ws_size,
                              hipStream_t stream) {
  const float* x  = (const float*)d_in[0];
  const float* Wq = (const float*)d_in[1];
  const float* bq = (const float*)d_in[2];
  const float* Wk = (const float*)d_in[3];
  const float* bk = (const float*)d_in[4];
  const float* Wv = (const float*)d_in[5];
  const float* bv = (const float*)d_in[6];
  const float* Wo = (const float*)d_in[7];
  const float* bo = (const float*)d_in[8];
  float* out = (float*)d_out;

  // workspace carve-up (all slices 256B aligned; total ~88 MB)
  constexpr size_t N_X    = (size_t)NROWS * DMODEL;        // 8M elems
  constexpr size_t N_W    = (size_t)DMODEL * DMODEL;       // 1M elems
  constexpr size_t N_HEAD = (size_t)BATCH * NHEADS * SEQ * DK; // 8M elems
  char* ws = (char*)d_ws;
  _Float16* Xh   = (_Float16*)ws;                 ws += N_X * 2;
  _Float16* Wqkv = (_Float16*)ws;                 ws += 3 * N_W * 2;
  _Float16* Woh  = (_Float16*)ws;                 ws += N_W * 2;
  _Float16* Qh   = (_Float16*)ws;                 ws += N_HEAD * 2;
  _Float16* Kh   = (_Float16*)ws;                 ws += N_HEAD * 2;
  _Float16* Vt   = (_Float16*)ws;                 ws += N_HEAD * 2;
  _Float16* Ctx  = (_Float16*)ws;                 ws += N_X * 2;

  // 0) f32 -> f16 conversions (x, packed Wqkv, Wo)
  {
    int n = (int)N_X;
    cvt_f32_f16<<<(n + 255) / 256, 256, 0, stream>>>(x, Xh, n);
    n = (int)N_W;
    cvt_f32_f16<<<(n + 255) / 256, 256, 0, stream>>>(Wq, Wqkv,             n);
    cvt_f32_f16<<<(n + 255) / 256, 256, 0, stream>>>(Wk, Wqkv + N_W,       n);
    cvt_f32_f16<<<(n + 255) / 256, 256, 0, stream>>>(Wv, Wqkv + 2 * N_W,   n);
    cvt_f32_f16<<<(n + 255) / 256, 256, 0, stream>>>(Wo, Woh,              n);
  }

  // 1) fused QKV projection (32-row tiles per wave)
  {
    dim3 grid(NROWS / 256, QKV_N / 64);
    qkv_gemm<<<grid, 256, 0, stream>>>(Xh, Wqkv, bq, bk, bv, Qh, Kh, Vt);
  }

  // 2) flash attention
  {
    dim3 grid(SEQ / 128, BATCH * NHEADS);
    attn_kernel<<<grid, 256, 0, stream>>>(Qh, Kh, Vt, Ctx);
  }

  // 3) output projection (32-row tiles per wave)
  {
    dim3 grid(NROWS / 256, DMODEL / 64);
    out_gemm<<<grid, 256, 0, stream>>>(Ctx, Woh, bo, out);
  }
}
